// SplineNet_85538568667548
// MI455X (gfx1250) — matrix operand order repaired and checked by reading, not compiled, not run on hardware
//
#include <hip/hip_runtime.h>
#include <hip/hip_bf16.h>
#include <stdint.h>

typedef __attribute__((ext_vector_type(16))) __bf16 v16bf;
typedef __attribute__((ext_vector_type(8)))  float  v8f;

// ---- helpers ---------------------------------------------------------------

static __device__ __forceinline__ __bf16 f2bf(float f) {
  unsigned u = __float_as_uint(f);
  u = (u + 0x7FFFu + ((u >> 16) & 1u)) >> 16;   // round-to-nearest-even
  unsigned short s = (unsigned short)u;
  return __builtin_bit_cast(__bf16, s);
}

// ---- small utility kernels -------------------------------------------------

__global__ void zero_i32_kernel(int* __restrict__ p, int n) {
  int i = blockIdx.x * blockDim.x + threadIdx.x;
  if (i < n) p[i] = 0;
}

__global__ void copy_i32_kernel(int* __restrict__ dst, const int* __restrict__ src, int n) {
  int i = blockIdx.x * blockDim.x + threadIdx.x;
  if (i < n) dst[i] = src[i];
}

// dst[o*stride + coloff + i] = bf16(src[i*OUT + o])   (transpose + convert)
__global__ void pack_wt_kernel(const float* __restrict__ src, __bf16* __restrict__ dstp,
                               int IN, int OUT, int stride, int coloff) {
  int t = blockIdx.x * blockDim.x + threadIdx.x;
  if (t >= IN * OUT) return;
  int i = t / OUT, o = t - i * OUT;
  dstp[(long long)o * stride + coloff + i] = f2bf(src[(long long)i * OUT + o]);
}

// ---- CSR build (edge_index is identical for both conv layers) --------------

__global__ void csr_count_kernel(const int* __restrict__ dst, int* __restrict__ cnt, int E) {
  int e = blockIdx.x * blockDim.x + threadIdx.x;
  if (e < E) atomicAdd(&cnt[dst[e]], 1);
}

// Single-block exclusive scan: rowptr[0..N] from cnt[0..N-1].
__global__ void csr_scan_kernel(const int* __restrict__ cnt, int* __restrict__ rowptr, int N) {
  __shared__ int part[1024];
  int t = threadIdx.x;
  int CH = (N + 1023) >> 10;
  int beg = t * CH;
  int end = beg + CH < N ? beg + CH : N;
  int s = 0;
  for (int i = beg; i < end; ++i) s += cnt[i];
  part[t] = s;
  __syncthreads();
  for (int off = 1; off < 1024; off <<= 1) {          // Hillis-Steele inclusive scan
    int add = (t >= off) ? part[t - off] : 0;
    __syncthreads();
    part[t] += add;
    __syncthreads();
  }
  int prefix = part[t] - s;                            // exclusive prefix for this chunk
  for (int i = beg; i < end; ++i) { rowptr[i] = prefix; prefix += cnt[i]; }
  if (t == 1023) rowptr[N] = part[1023];
}

__global__ void csr_fill_kernel(const int* __restrict__ dst, int* __restrict__ cursor,
                                int* __restrict__ eidx, int E) {
  int e = blockIdx.x * blockDim.x + threadIdx.x;
  if (e >= E) return;
  int pos = atomicAdd(&cursor[dst[e]], 1);
  eidx[pos] = e;
}

// ---- pull-mode spline aggregation (no float atomics) -----------------------
// One wave per node. Lane L accumulates features L and L+32 for all K slots.
// Edge meta (src, attr) is read once per edge by one lane and shfl-broadcast.
template <int K>
__global__ void aggregate_kernel(const float* __restrict__ xin, const int* __restrict__ srcIdx,
                                 const float* __restrict__ attr, const int* __restrict__ rowptr,
                                 const int* __restrict__ eidx, float* __restrict__ agg, int n) {
  int node = blockIdx.x * (blockDim.x >> 5) + (threadIdx.x >> 5);
  if (node >= n) return;
  int lane = threadIdx.x & 31;
  int beg = rowptr[node], end = rowptr[node + 1];

  float acc0[K], acc1[K];
#pragma unroll
  for (int k = 0; k < K; ++k) { acc0[k] = 0.0f; acc1[k] = 0.0f; }

  for (int base = beg; base < end; base += 32) {
    int m = end - base;
    int cntc = m < 32 ? m : 32;
    int mys = 0; float myp = 0.0f;
    if (lane < cntc) {
      int ee = eidx[base + lane];
      mys = srcIdx[ee];
      myp = attr[ee];
    }
    for (int i = 0; i < cntc; ++i) {
      int   s = __shfl(mys, i);
      float p = __shfl(myp, i);
      float v = p * (float)(K - 1);
      float fl = floorf(v);
      float frac = v - fl;
      int i0 = (int)fl;
      i0 = i0 < 0 ? 0 : (i0 > K - 1 ? K - 1 : i0);
      int i1 = (i0 + 1 > K - 1) ? (K - 1) : (i0 + 1);
      float xv0 = xin[(long long)s * 64 + lane];
      float xv1 = xin[(long long)s * 64 + 32 + lane];
      float c0 = 1.0f - frac;
#pragma unroll
      for (int k = 0; k < K; ++k) {
        float ck = (k == i0 ? c0 : 0.0f) + (k == i1 ? frac : 0.0f);
        acc0[k] += ck * xv0;
        acc1[k] += ck * xv1;
      }
    }
  }
#pragma unroll
  for (int k = 0; k < K; ++k) {
    float* dstrow = agg + ((long long)k * n + node) * 64;
    dstrow[lane]      = acc0[k];
    dstrow[lane + 32] = acc1[k];
  }
}

// ---- WMMA GEMM + activation ------------------------------------------------
// One wave -> 16-row x (NTILES*16)-col strip of
//   out = act( [agg_0/deg | ... | agg_{K-1}/deg | xsrc] @ Wbig + bias )
// Weights staged in LDS (row-padded by 8 bf16 -> conflict-free B-fragment reads).
// ACT: 0 = ELU, 1 = ReLU.
template <int NTILES, int ACT>
__global__ void gemm_act_kernel(const float* __restrict__ agg, int K,
                                const int* __restrict__ rowptr,
                                const float* __restrict__ xsrc,
                                const __bf16* __restrict__ Wt,   // [NCOLS, Ktot]
                                const float* __restrict__ bias,
                                float* __restrict__ out, int nrows) {
  const int Ktot  = (K + 1) * 64;
  const int NCOLS = NTILES * 16;
  const int LSTR  = Ktot + 8;                  // padded LDS row stride (bf16 elems)

  extern __shared__ char smem_raw[];
  __bf16* Ws = (__bf16*)smem_raw;

  // Cooperative stage of Wt into LDS (uint4 chunks, row-wise to apply padding).
  const int rowvecs = Ktot >> 3;               // uint4 per source row
  for (int i = threadIdx.x; i < NCOLS * rowvecs; i += blockDim.x) {
    int o = i / rowvecs, cv = i - o * rowvecs;
    ((uint4*)(Ws + o * LSTR))[cv] = ((const uint4*)(Wt + (long long)o * Ktot))[cv];
  }
  __syncthreads();

  int wave = blockIdx.x * (blockDim.x >> 5) + (threadIdx.x >> 5);
  int row0 = wave * 16;
  if (row0 >= nrows) return;                   // wave-uniform; after the only barrier
  int lane = threadIdx.x & 31;
  int hf   = lane >> 4;
  int l15  = lane & 15;
  int rowA = row0 + l15;

  float invd = 1.0f;
  if (K > 0) {
    int d = rowptr[rowA + 1] - rowptr[rowA];
    invd = 1.0f / (float)(d > 1 ? d : 1);
  }

  v8f c[NTILES] = {};

  for (int kk = 0; kk < Ktot; kk += 32) {
    int blk = kk >> 6;
    const float* arow;
    float scale;
    if (blk < K) { arow = agg + ((long long)blk * nrows + rowA) * 64; scale = invd; }
    else         { arow = xsrc + (long long)rowA * 64;                scale = 1.0f; }
    // A 16x32 bf16 layout: elems 0..7 <- K = kk + hf*8 .. +7 ; elems 8..15 <- +16
    int k0 = (kk & 63) + hf * 8;
    const float4* p0 = (const float4*)(arow + k0);
    const float4* p1 = (const float4*)(arow + k0 + 16);
    float4 f0 = p0[0], f1 = p0[1], f2 = p1[0], f3 = p1[1];
    v16bf a;
    a[0]=f2bf(f0.x*scale);  a[1]=f2bf(f0.y*scale);  a[2]=f2bf(f0.z*scale);  a[3]=f2bf(f0.w*scale);
    a[4]=f2bf(f1.x*scale);  a[5]=f2bf(f1.y*scale);  a[6]=f2bf(f1.z*scale);  a[7]=f2bf(f1.w*scale);
    a[8]=f2bf(f2.x*scale);  a[9]=f2bf(f2.y*scale);  a[10]=f2bf(f2.z*scale); a[11]=f2bf(f2.w*scale);
    a[12]=f2bf(f3.x*scale); a[13]=f2bf(f3.y*scale); a[14]=f2bf(f3.z*scale); a[15]=f2bf(f3.w*scale);
#pragma unroll
    for (int t = 0; t < NTILES; ++t) {
      int outcol = t * 16 + l15;
      // B 32x16 bf16: elems 0..7 <- Ws[outcol, kk+hf*8 .. +7] ; elems 8..15 <- +16
      const __bf16* wrow = Ws + outcol * LSTR + kk + hf * 8;
      union { uint4 q[2]; v16bf v; } bu;
      bu.q[0] = *(const uint4*)wrow;
      bu.q[1] = *(const uint4*)(wrow + 16);
      c[t] = __builtin_amdgcn_wmma_f32_16x16x32_bf16(false, a, false, bu.v,
                                                     (short)0, c[t], false, false);
    }
  }

#pragma unroll
  for (int t = 0; t < NTILES; ++t) {
    int outcol = t * 16 + l15;
    float bv = bias[outcol];
#pragma unroll
    for (int r = 0; r < 8; ++r) {
      int row = row0 + r + 8 * hf;            // C/D layout: VGPR r -> row r + 8*half
      float v = c[t][r] + bv;
      if (ACT == 0) v = (v > 0.0f) ? v : (expf(v) - 1.0f);   // ELU
      else          v = fmaxf(v, 0.0f);                      // ReLU
      if (row < nrows) out[(long long)row * NCOLS + outcol] = v;
    }
  }
}

// ---- launch ----------------------------------------------------------------

extern "C" void kernel_launch(void* const* d_in, const int* in_sizes, int n_in,
                              void* d_out, int out_size, void* d_ws, size_t ws_size,
                              hipStream_t stream) {
  const float* x     = (const float*)d_in[0];
  const int*   ei    = (const int*)d_in[1];
  const float* attr  = (const float*)d_in[2];
  const float* W1    = (const float*)d_in[3];
  const float* root1 = (const float*)d_in[4];
  const float* b1    = (const float*)d_in[5];
  const float* W2    = (const float*)d_in[6];
  const float* root2 = (const float*)d_in[7];
  const float* b2    = (const float*)d_in[8];
  const float* Wm1   = (const float*)d_in[9];
  const float* bm1   = (const float*)d_in[10];
  const float* Wm2   = (const float*)d_in[11];
  const float* bm2   = (const float*)d_in[12];

  const int N = in_sizes[0] / 64;   // 100000
  const int E = in_sizes[1] / 2;    // 1600000
  const int* srcIdx = ei;           // edge_index[0,:]
  const int* dstIdx = ei + E;       // edge_index[1,:]

  char* ws = (char*)d_ws;
  size_t off = 0;
  auto take = [&](size_t bytes) -> char* {
    char* p = ws + off;
    off = (off + bytes + 255) & ~(size_t)255;
    return p;
  };
  __bf16* wt1    = (__bf16*)take((size_t)64 * 256 * 2);   // [64,256] = [W1(3);root1]^T
  __bf16* wt2    = (__bf16*)take((size_t)64 * 384 * 2);   // [64,384] = [W2(5);root2]^T
  __bf16* wm1t   = (__bf16*)take((size_t)64 * 64 * 2);
  __bf16* wm2t   = (__bf16*)take((size_t)16 * 64 * 2);
  int*    cnt    = (int*)take((size_t)N * 4);
  int*    rowptr = (int*)take((size_t)(N + 1) * 4);
  int*    cursor = (int*)take((size_t)N * 4);
  int*    eidx   = (int*)take((size_t)E * 4);
  float*  agg    = (float*)take((size_t)5 * N * 64 * 4);  // reused between layers
  float*  h1     = (float*)take((size_t)N * 64 * 4);
  float*  h2     = (float*)take((size_t)N * 64 * 4);
  float*  h3     = (float*)take((size_t)N * 64 * 4);
  (void)ws_size; (void)n_in; (void)out_size;

  dim3 blk(256);

  // Pack weights (bf16, transposed so B-fragments are contiguous per output col)
  pack_wt_kernel<<<(192 * 64 + 255) / 256, blk, 0, stream>>>(W1,    wt1, 192, 64, 256, 0);
  pack_wt_kernel<<<(64  * 64 + 255) / 256, blk, 0, stream>>>(root1, wt1, 64,  64, 256, 192);
  pack_wt_kernel<<<(320 * 64 + 255) / 256, blk, 0, stream>>>(W2,    wt2, 320, 64, 384, 0);
  pack_wt_kernel<<<(64  * 64 + 255) / 256, blk, 0, stream>>>(root2, wt2, 64,  64, 384, 320);
  pack_wt_kernel<<<(64  * 64 + 255) / 256, blk, 0, stream>>>(Wm1,   wm1t, 64, 64, 64, 0);
  pack_wt_kernel<<<(64  * 16 + 255) / 256, blk, 0, stream>>>(Wm2,   wm2t, 64, 16, 64, 0);

  // CSR of incoming edges (shared by both conv layers; also provides degree)
  zero_i32_kernel<<<(N + 255) / 256, blk, 0, stream>>>(cnt, N);
  csr_count_kernel<<<(E + 255) / 256, blk, 0, stream>>>(dstIdx, cnt, E);
  csr_scan_kernel<<<1, 1024, 0, stream>>>(cnt, rowptr, N);
  copy_i32_kernel<<<(N + 255) / 256, blk, 0, stream>>>(cursor, rowptr, N);
  csr_fill_kernel<<<(E + 255) / 256, blk, 0, stream>>>(dstIdx, cursor, eidx, E);

  const int AGG_BLOCKS  = (N + 7) / 8;                  // 8 waves (nodes) / block
  const int WAVES       = (N + 15) / 16;
  const int GEMM_BLOCKS = (WAVES + 7) / 8;              // 8 waves / 256-thread block
  const size_t SH1 = (size_t)64 * (256 + 8) * 2;        // LDS for staged weights
  const size_t SH2 = (size_t)64 * (384 + 8) * 2;
  const size_t SH3 = (size_t)64 * (64 + 8) * 2;
  const size_t SH4 = (size_t)16 * (64 + 8) * 2;

  // SplineConv layer 1 (K=3) + ELU
  aggregate_kernel<3><<<AGG_BLOCKS, blk, 0, stream>>>(x, srcIdx, attr, rowptr, eidx, agg, N);
  gemm_act_kernel<4, 0><<<GEMM_BLOCKS, blk, SH1, stream>>>(agg, 3, rowptr, x, wt1, b1, h1, N);

  // SplineConv layer 2 (K=5) + ELU
  aggregate_kernel<5><<<AGG_BLOCKS, blk, 0, stream>>>(h1, srcIdx, attr, rowptr, eidx, agg, N);
  gemm_act_kernel<4, 0><<<GEMM_BLOCKS, blk, SH2, stream>>>(agg, 5, rowptr, h1, wt2, b2, h2, N);

  // MLP head: ReLU(h2@Wm1+bm1) then ReLU(@Wm2+bm2) -> d_out [N,16]
  gemm_act_kernel<4, 1><<<GEMM_BLOCKS, blk, SH3, stream>>>(nullptr, 0, nullptr, h2, wm1t, bm1, h3, N);
  gemm_act_kernel<1, 1><<<GEMM_BLOCKS, blk, SH4, stream>>>(nullptr, 0, nullptr, h3, wm2t, bm2,
                                                           (float*)d_out, N);
}
